// Attention_80229989089713
// MI455X (gfx1250) — compile-verified, matrix operand order checked by reading, not run on hardware
//
#include <hip/hip_runtime.h>
#include <hip/hip_bf16.h>
#include <stddef.h>

// ---------------------------------------------------------------------------
// MI455X (gfx1250): wave32, v_wmma_f32_16x16x32_bf16, fragment-swizzled LDS
// (each fragment = one 32-byte per-lane vector load -> 2x ds_load_b128),
// software-pipelined B-fragment loads, double-buffered GEMM staging.
//
// Fragment element mapping (ISA 7.12.2, 16-bit A/B 16x32):
//   within-chunk k (0..31):  h = (k>>3)&1,  e = (k&7) + ((k>>4)<<3)
//   lane = (rowOrCol & 15) + 16*h, element e in the lane's 16-bf16 vector.
//
// Workspace (floats): P,Q,Lc,G [65536x128] + Mfuse[128x512] + Wfsi[128x128]
// (~128.3 MiB). All intermediates fit in the 192 MB L2.
// ---------------------------------------------------------------------------

#define Bb 16
#define Tt 4096
#define Ee 512
#define Dd 128
#define MTOT (Bb * Tt)
#define LOCAL_STEPS 64
#define SUMM_MASK 31
#define LN_EPS 1e-5f

typedef __attribute__((ext_vector_type(16))) __bf16 v16bf;
typedef __attribute__((ext_vector_type(4)))  __bf16 v4bf;
typedef __attribute__((ext_vector_type(8)))  float  v8f;
typedef __attribute__((ext_vector_type(4)))  float  v4f;

__device__ __forceinline__ v8f wmma_bf16(v16bf a, v16bf b, v8f c) {
  return __builtin_amdgcn_wmma_f32_16x16x32_bf16(
      false, a, false, b, (short)0, c, false, false);
}

__device__ __forceinline__ v16bf load_frag(const __bf16* base) {
  return *(const v16bf*)(base + (threadIdx.x & 31) * 16);
}

__device__ __forceinline__ int swz_h(int kn) { return (kn >> 3) & 1; }
__device__ __forceinline__ int swz_e(int kn) { return (kn & 7) + ((kn >> 4) << 3); }

// 8 N-tiles of WMMA against one A fragment, B frags pipelined one tile ahead.
__device__ __forceinline__ void wmma_row8(v8f* acc, v16bf af,
                                          const __bf16* bbase) {
  v16bf b0 = load_frag(bbase);
#pragma unroll
  for (int tile = 0; tile < 8; ++tile) {
    v16bf b1 = (tile < 7) ? load_frag(bbase + (tile + 1) * 512) : b0;
    acc[tile] = wmma_bf16(af, b0, acc[tile]);
    b0 = b1;
  }
}

// Stage a 128x128 f32 weight [k][n] into fragment-swizzled bf16 LDS:
// chunk = (k>>5)*8 + (n>>4); 32 chunks * 32 lanes * 16 elems = 16384 bf16.
__device__ __forceinline__ void stage_weight_swz(const float* __restrict__ W,
                                                 __bf16* dst, int tid,
                                                 int nthreads) {
  for (int i = tid; i < 128 * 128; i += nthreads) {
    int k = i >> 7, n = i & 127;
    int kn = k & 31;
    int idx = ((((k >> 5) << 3) + (n >> 4)) * 32 + (n & 15) + 16 * swz_h(kn)) * 16 +
              swz_e(kn);
    dst[idx] = (__bf16)W[i];
  }
}

// ---------------------------------------------------------------------------
// Streaming GEMM: C[M,N] = A[M,K] @ Bw[K,N] (+bias). 128x128 tile, 8 waves,
// wave w owns rows [w*16, w*16+16) x all 8 N-tiles. K chunks of 32 with
// double-buffered LDS staging (stage chunk c+1 while computing chunk c).
// ---------------------------------------------------------------------------
__device__ __forceinline__ void gemm_stage_chunk(
    const float* __restrict__ A, const float* __restrict__ Bw,
    __bf16* a_swz, __bf16* b_swz, int m0, int n0, int k0, int K, int N,
    int tid) {
  // A chunk: 128 rows x 32 k, float4 loads; a 4-aligned k-group maps to 4
  // contiguous bf16 in one lane's fragment (one b64 LDS store).
  for (int i = tid; i < 128 * 8; i += 256) {
    int r = i >> 3, c0 = (i & 7) << 2;
    v4f x4 = *(const v4f*)&A[(size_t)(m0 + r) * K + k0 + c0];
    int hh = swz_h(c0), e0 = swz_e(c0);
    v4bf y;
    y[0] = (__bf16)x4[0]; y[1] = (__bf16)x4[1];
    y[2] = (__bf16)x4[2]; y[3] = (__bf16)x4[3];
    *(v4bf*)&a_swz[((((r >> 4) << 5) + (r & 15) + 16 * hh) << 4) + e0] = y;
  }
  // B chunk: 32 k x 128 n, float4 along n; scattered b16 stores (lane=f(n)).
  for (int i = tid; i < 32 * 32; i += 256) {
    int r = i >> 5, c0 = (i & 31) << 2;  // k = r, n = c0..c0+3
    v4f x4 = *(const v4f*)&Bw[(size_t)(k0 + r) * N + n0 + c0];
    int hh = swz_h(r), e = swz_e(r);
#pragma unroll
    for (int u = 0; u < 4; ++u) {
      int n = c0 + u;
      b_swz[((((n >> 4) << 5) + (n & 15) + 16 * hh) << 4) + e] = (__bf16)x4[u];
    }
  }
}

__global__ __launch_bounds__(256) void gemm_bf16_tiled(
    const float* __restrict__ A, const float* __restrict__ Bw,
    const float* __restrict__ bias, float* __restrict__ C,
    int M, int N, int K) {
  __shared__ __align__(32) __bf16 a_swz[2][8 * 32 * 16];
  __shared__ __align__(32) __bf16 b_swz[2][8 * 32 * 16];
  const int tid = threadIdx.x;
  const int wave = tid >> 5, ln = tid & 15, h = (tid >> 4) & 1;
  const int m0 = blockIdx.x * 128, n0 = blockIdx.y * 128;
  const int nchunks = K >> 5;

  v8f acc[8] = {};
  gemm_stage_chunk(A, Bw, a_swz[0], b_swz[0], m0, n0, 0, K, N, tid);
  __syncthreads();
  for (int c = 0; c < nchunks; ++c) {
    int cur = c & 1;
    if (c + 1 < nchunks)
      gemm_stage_chunk(A, Bw, a_swz[cur ^ 1], b_swz[cur ^ 1], m0, n0,
                       (c + 1) << 5, K, N, tid);
    v16bf af = load_frag(&a_swz[cur][wave * 512]);
    wmma_row8(acc, af, &b_swz[cur][0]);
    __syncthreads();
  }
#pragma unroll
  for (int tile = 0; tile < 8; ++tile) {
    int n = n0 + tile * 16 + ln;
    float bv = (bias != nullptr) ? bias[n] : 0.0f;
#pragma unroll
    for (int v = 0; v < 8; ++v) {
      int row = m0 + wave * 16 + v + 8 * h;
      C[(size_t)row * N + n] = acc[tile][v] + bv;
    }
  }
}

// ---------------------------------------------------------------------------
// Weight prefolds: Mfuse[d,n] = sum_e Go[d,e]*linw[n,e]   (128x512)
//                  Wfsi [d,n] = sum_e Go[d,e]*Wsi[e,n]    (128x128)
// ---------------------------------------------------------------------------
__global__ __launch_bounds__(256) void fuse_weights_kernel(
    const float* __restrict__ Go, const float* __restrict__ linw,
    const float* __restrict__ Wsi, float* __restrict__ Mfuse,
    float* __restrict__ Wfsi) {
  int idx = blockIdx.x * blockDim.x + threadIdx.x;
  if (idx < 128 * 512) {
    int d = idx >> 9, n = idx & 511;
    float s = 0.0f;
    for (int e = 0; e < 512; ++e) s += Go[d * 512 + e] * linw[n * 512 + e];
    Mfuse[idx] = s;
  } else if (idx < 128 * 512 + 128 * 128) {
    int j = idx - 128 * 512;
    int d = j >> 7, n = j & 127;
    float s = 0.0f;
    for (int e = 0; e < 512; ++e) s += Go[d * 512 + e] * Wsi[e * 128 + n];
    Wfsi[j] = s;
  }
}

// ---------------------------------------------------------------------------
// Local windowed recurrence: 4 waves/WG, wave owns 16 t's of one batch; S tile
// [16x128] lives in f32 accumulators across all 64 window steps. Per step:
// Sreg = LN(S@W_ls + P[t-1-j]); rows with t<=j frozen. Then S@blend_shaper.
// ---------------------------------------------------------------------------
__global__ __launch_bounds__(128) void local_recurrence_kernel(
    const float* __restrict__ P, const float* __restrict__ Wls,
    const float* __restrict__ Wblend, const float* __restrict__ gamma,
    const float* __restrict__ beta, float* __restrict__ Lc) {
  __shared__ __align__(32) __bf16 w_swz[32 * 32 * 16];     // [kk*8+tile][lane][16]
  __shared__ __align__(32) __bf16 s_swz[4][4 * 32 * 16];   // per wave [kk][lane][16]

  const int tid = threadIdx.x;
  const int wave = tid >> 5, lane = tid & 31, ln = tid & 15, h = (tid >> 4) & 1;
  const int hb = (ln >> 3) & 1, eb = ln & 7;  // S-publish swizzle from ln
  const int g0 = blockIdx.x * 64;
  const int b = g0 >> 12;
  const int t0w = (g0 & (Tt - 1)) + wave * 16;
  const int bT = b << 12;
  __bf16* s_w = &s_swz[wave][0];

  stage_weight_swz(Wls, w_swz, tid, 128);
  for (int i = lane; i < 4 * 32 * 16; i += 32) s_w[i] = (__bf16)0.0f;

  float gam[8], bet[8];
#pragma unroll
  for (int tile = 0; tile < 8; ++tile) {
    gam[tile] = gamma[tile * 16 + ln];
    bet[tile] = beta[tile * 16 + ln];
  }
  v8f Sreg[8] = {};
  __syncthreads();

  for (int j = 0; j < LOCAL_STEPS; ++j) {
    v8f acc[8];
#pragma unroll
    for (int tile = 0; tile < 8; ++tile) {
      int n = tile * 16 + ln;
#pragma unroll
      for (int v = 0; v < 8; ++v) {
        int src = t0w + v + 8 * h - 1 - j;
        acc[tile][v] = (src >= 0) ? P[(size_t)(bT + src) * 128 + n] : 0.0f;
      }
    }
#pragma unroll
    for (int kk = 0; kk < 4; ++kk)
      wmma_row8(acc, load_frag(&s_w[kk * 512]), &w_swz[(kk * 8) * 512]);
#pragma unroll
    for (int v = 0; v < 8; ++v) {
      float s = 0.0f, q = 0.0f;
#pragma unroll
      for (int tile = 0; tile < 8; ++tile) {
        float x = acc[tile][v];
        s += x; q += x * x;
      }
#pragma unroll
      for (int msk = 1; msk < 16; msk <<= 1) {
        s += __shfl_xor(s, msk, 32);
        q += __shfl_xor(q, msk, 32);
      }
      float mean = s * (1.0f / 128.0f);
      float var = q * (1.0f / 128.0f) - mean * mean;
      float rstd = rsqrtf(var + LN_EPS);
      bool valid = (t0w + v + 8 * h) >= (j + 1);
#pragma unroll
      for (int tile = 0; tile < 8; ++tile) {
        float val = (acc[tile][v] - mean) * rstd * gam[tile] + bet[tile];
        if (valid) Sreg[tile][v] = val;
      }
    }
    __syncthreads();
    // publish S directly in A-fragment swizzle: (m = v+8h, n = tile*16+ln)
#pragma unroll
    for (int tile = 0; tile < 8; ++tile)
#pragma unroll
      for (int v = 0; v < 8; ++v)
        s_w[((tile >> 1) << 9) + ((v + 8 * h + 16 * hb) << 4) + eb +
            ((tile & 1) << 3)] = (__bf16)Sreg[tile][v];
    __syncthreads();
  }

  // blended = S @ local_blend_shaper
  stage_weight_swz(Wblend, w_swz, tid, 128);
  __syncthreads();
  v8f accb[8] = {};
#pragma unroll
  for (int kk = 0; kk < 4; ++kk)
    wmma_row8(accb, load_frag(&s_w[kk * 512]), &w_swz[(kk * 8) * 512]);
#pragma unroll
  for (int tile = 0; tile < 8; ++tile) {
    int n = tile * 16 + ln;
#pragma unroll
    for (int v = 0; v < 8; ++v)
      Lc[(size_t)(bT + t0w + v + 8 * h) * 128 + n] = accb[tile][v];
  }
}

// ---------------------------------------------------------------------------
// Global sequential recurrence: 1 WG (4 waves), 4096 dependent steps.
// g/summ/tmp live as swizzled bf16 fragment tiles (written by LN epilogue);
// summ additionally keeps an f32 copy for the additive term. Wave owns 32
// columns (2 N-tiles); cross-wave LN via shuffle butterfly + LDS reduce.
// ---------------------------------------------------------------------------
__global__ __launch_bounds__(128) void global_recurrence_kernel(
    const float* __restrict__ Q, const float* __restrict__ Lc,
    const float* __restrict__ Wg, const float* __restrict__ Wsc,
    const float* __restrict__ Wfsi, const float* __restrict__ Wso,
    const float* __restrict__ gamma, const float* __restrict__ beta,
    float* __restrict__ G) {
  extern __shared__ __align__(32) char smem[];
  __bf16* wg_w   = (__bf16*)smem;            // each 32*32*16 = 16384 bf16
  __bf16* wsc_w  = wg_w + 16384;
  __bf16* wfsi_w = wsc_w + 16384;
  __bf16* wso_w  = wfsi_w + 16384;
  __bf16* g_swz  = wso_w + 16384;            // each 4*32*16 = 2048 bf16
  __bf16* sm_swz = g_swz + 2048;
  __bf16* tmp_swz= sm_swz + 2048;
  float* sm_f32  = (float*)(tmp_swz + 2048); // 16 x 132
  float* redS    = sm_f32 + 16 * 132;        // [4 waves][16 rows]
  float* redQ    = redS + 64;

  const int tid = threadIdx.x;
  const int wave = tid >> 5, ln = tid & 15, h = (tid >> 4) & 1;
  const int hb = (ln >> 3) & 1, eb = ln & 7;

  stage_weight_swz(Wg,   wg_w,   tid, 128);
  stage_weight_swz(Wsc,  wsc_w,  tid, 128);
  stage_weight_swz(Wfsi, wfsi_w, tid, 128);
  stage_weight_swz(Wso,  wso_w,  tid, 128);
  for (int i = tid; i < 2048; i += 128) {
    g_swz[i] = (__bf16)0.0f; sm_swz[i] = (__bf16)0.0f; tmp_swz[i] = (__bf16)0.0f;
  }
  for (int i = tid; i < 16 * 132; i += 128) sm_f32[i] = 0.0f;

  float gamv[2], betv[2];
#pragma unroll
  for (int tl = 0; tl < 2; ++tl) {
    int n = wave * 32 + tl * 16 + ln;
    gamv[tl] = gamma[n]; betv[tl] = beta[n];
  }
  __syncthreads();

  for (int t = 0; t < Tt; ++t) {
    v16bf ag[4];
#pragma unroll
    for (int kk = 0; kk < 4; ++kk) ag[kk] = load_frag(&g_swz[kk * 512]);
    __syncthreads();

    v8f acc[2];
#pragma unroll
    for (int tl = 0; tl < 2; ++tl) {
      int n = wave * 32 + tl * 16 + ln;
#pragma unroll
      for (int v = 0; v < 8; ++v) {
        int m = v + 8 * h;
        size_t r = (size_t)m * Tt + t;
        acc[tl][v] = Q[r * 128 + n] + Lc[r * 128 + n] + sm_f32[m * 132 + n];
      }
#pragma unroll
      for (int kk = 0; kk < 4; ++kk)
        acc[tl] = wmma_bf16(ag[kk],
                            load_frag(&wg_w[(kk * 8 + wave * 2 + tl) * 512]),
                            acc[tl]);
    }
#pragma unroll
    for (int v = 0; v < 8; ++v) {
      float s = acc[0][v] + acc[1][v];
      float q = acc[0][v] * acc[0][v] + acc[1][v] * acc[1][v];
#pragma unroll
      for (int msk = 1; msk < 16; msk <<= 1) {
        s += __shfl_xor(s, msk, 32);
        q += __shfl_xor(q, msk, 32);
      }
      if (ln == 0) {
        int m = v + 8 * h;
        redS[wave * 16 + m] = s;
        redQ[wave * 16 + m] = q;
      }
    }
    __syncthreads();
#pragma unroll
    for (int v = 0; v < 8; ++v) {
      int m = v + 8 * h;
      float s = redS[m] + redS[16 + m] + redS[32 + m] + redS[48 + m];
      float q = redQ[m] + redQ[16 + m] + redQ[32 + m] + redQ[48 + m];
      float mean = s * (1.0f / 128.0f);
      float var = q * (1.0f / 128.0f) - mean * mean;
      float rstd = rsqrtf(var + LN_EPS);
#pragma unroll
      for (int tl = 0; tl < 2; ++tl) {
        int n = wave * 32 + tl * 16 + ln;
        float val = (acc[tl][v] - mean) * rstd * gamv[tl] + betv[tl];
        g_swz[((wave * 32 + m + 16 * hb) << 4) + eb + (tl << 3)] = (__bf16)val;
        G[((size_t)m * Tt + t) * 128 + n] = val;
      }
    }

    if ((t & SUMM_MASK) == SUMM_MASK) {
      __syncthreads();
      v16bf as[4], ag2[4];
#pragma unroll
      for (int kk = 0; kk < 4; ++kk) {
        as[kk]  = load_frag(&sm_swz[kk * 512]);
        ag2[kk] = load_frag(&g_swz[kk * 512]);
      }
      __syncthreads();
      v8f a2[2] = {};
#pragma unroll
      for (int tl = 0; tl < 2; ++tl) {
        int nb = wave * 2 + tl;
#pragma unroll
        for (int kk = 0; kk < 4; ++kk) {
          a2[tl] = wmma_bf16(as[kk],  load_frag(&wsc_w[(kk * 8 + nb) * 512]), a2[tl]);
          a2[tl] = wmma_bf16(ag2[kk], load_frag(&wfsi_w[(kk * 8 + nb) * 512]), a2[tl]);
        }
      }
#pragma unroll
      for (int v = 0; v < 8; ++v) {
        float s = a2[0][v] + a2[1][v];
        float q = a2[0][v] * a2[0][v] + a2[1][v] * a2[1][v];
#pragma unroll
        for (int msk = 1; msk < 16; msk <<= 1) {
          s += __shfl_xor(s, msk, 32);
          q += __shfl_xor(q, msk, 32);
        }
        if (ln == 0) {
          int m = v + 8 * h;
          redS[wave * 16 + m] = s;
          redQ[wave * 16 + m] = q;
        }
      }
      __syncthreads();
#pragma unroll
      for (int v = 0; v < 8; ++v) {
        int m = v + 8 * h;
        float s = redS[m] + redS[16 + m] + redS[32 + m] + redS[48 + m];
        float q = redQ[m] + redQ[16 + m] + redQ[32 + m] + redQ[48 + m];
        float mean = s * (1.0f / 128.0f);
        float var = q * (1.0f / 128.0f) - mean * mean;
        float rstd = rsqrtf(var + LN_EPS);
#pragma unroll
        for (int tl = 0; tl < 2; ++tl) {
          float val = (a2[tl][v] - mean) * rstd * gamv[tl] + betv[tl];
          tmp_swz[((wave * 32 + m + 16 * hb) << 4) + eb + (tl << 3)] = (__bf16)val;
        }
      }
      __syncthreads();
      v16bf at[4];
#pragma unroll
      for (int kk = 0; kk < 4; ++kk) at[kk] = load_frag(&tmp_swz[kk * 512]);
      __syncthreads();
      v8f a3[2] = {};
#pragma unroll
      for (int tl = 0; tl < 2; ++tl) {
        int nb = wave * 2 + tl;
#pragma unroll
        for (int kk = 0; kk < 4; ++kk)
          a3[tl] = wmma_bf16(at[kk], load_frag(&wso_w[(kk * 8 + nb) * 512]), a3[tl]);
      }
#pragma unroll
      for (int tl = 0; tl < 2; ++tl) {
        int n = wave * 32 + tl * 16 + ln;
#pragma unroll
        for (int v = 0; v < 8; ++v) {
          int m = v + 8 * h;
          float val = a3[tl][v];
          sm_f32[m * 132 + n] = val;
          sm_swz[((wave * 32 + m + 16 * hb) << 4) + eb + (tl << 3)] = (__bf16)val;
        }
      }
    }
    __syncthreads();
  }
}

// ---------------------------------------------------------------------------
extern "C" void kernel_launch(void* const* d_in, const int* in_sizes, int n_in,
                              void* d_out, int out_size, void* d_ws,
                              size_t ws_size, hipStream_t stream) {
  (void)in_sizes; (void)n_in; (void)out_size; (void)ws_size;
  const float* x     = (const float*)d_in[0];
  const float* Wls   = (const float*)d_in[1];
  const float* Wli   = (const float*)d_in[2];
  const float* Wblend= (const float*)d_in[3];
  const float* Wsc   = (const float*)d_in[4];
  const float* Wsi   = (const float*)d_in[5];
  const float* Wso   = (const float*)d_in[6];
  const float* Wg    = (const float*)d_in[7];
  const float* Wgi   = (const float*)d_in[8];
  const float* Go    = (const float*)d_in[9];
  const float* gamma = (const float*)d_in[10];
  const float* beta  = (const float*)d_in[11];
  const float* linw  = (const float*)d_in[12];
  const float* linb  = (const float*)d_in[13];
  float* out = (float*)d_out;

  float* P    = (float*)d_ws;
  float* Qb   = P  + (size_t)MTOT * Dd;
  float* Lc   = Qb + (size_t)MTOT * Dd;
  float* G    = Lc + (size_t)MTOT * Dd;
  float* Mf   = G  + (size_t)MTOT * Dd;
  float* Wfsi = Mf + (size_t)Dd * Ee;

  gemm_bf16_tiled<<<dim3(MTOT / 128, 1), 256, 0, stream>>>(x, Wli, nullptr, P,
                                                           MTOT, Dd, Ee);
  gemm_bf16_tiled<<<dim3(MTOT / 128, 1), 256, 0, stream>>>(x, Wgi, nullptr, Qb,
                                                           MTOT, Dd, Ee);
  {
    int total = 128 * 512 + 128 * 128;
    fuse_weights_kernel<<<(total + 255) / 256, 256, 0, stream>>>(Go, linw, Wsi,
                                                                 Mf, Wfsi);
  }
  local_recurrence_kernel<<<MTOT / 64, 128, 0, stream>>>(P, Wls, Wblend, gamma,
                                                         beta, Lc);
  {
    const int smem3 = (4 * 16384 + 3 * 2048) * (int)sizeof(__bf16) +
                      (16 * 132 + 128) * (int)sizeof(float);
    (void)hipFuncSetAttribute((const void*)global_recurrence_kernel,
                              hipFuncAttributeMaxDynamicSharedMemorySize, smem3);
    global_recurrence_kernel<<<1, 128, smem3, stream>>>(Qb, Lc, Wg, Wsc, Wfsi,
                                                        Wso, gamma, beta, G);
  }
  gemm_bf16_tiled<<<dim3(MTOT / 128, Ee / 128), 256, 0, stream>>>(G, Mf, linb,
                                                                  out, MTOT,
                                                                  Ee, Dd);
}
